// GateInformedLinear_24275155157629
// MI455X (gfx1250) — compile-verified, matrix operand order checked by reading, not run on hardware
//
#include <hip/hip_runtime.h>
#include <hip/hip_bf16.h>

#define OUT_F 11008
#define IN_F  4096
#define KSPLIT 2
#define KHALF (IN_F / KSPLIT)

typedef __attribute__((ext_vector_type(16))) __bf16      v16bf;
typedef __attribute__((ext_vector_type(8)))  __bf16      v8bf;
typedef __attribute__((ext_vector_type(8)))  float       v8f;
typedef __attribute__((ext_vector_type(8)))  signed char v8i8;

static __device__ __forceinline__ v16bf cat16(v8bf a, v8bf b) {
    v16bf r;
#pragma unroll
    for (int i = 0; i < 8; ++i) { r[i] = a[i]; r[i + 8] = b[i]; }
    return r;
}

// Weights are streamed exactly once -> non-temporal load (TH=NT), keep L2 for x.
static __device__ __forceinline__ v8i8 ldnt8(const signed char* p) {
    return __builtin_nontemporal_load((const v8i8*)p);
}

// Low path : w = bf16(low * (low4_scale*scale))
// Full path: w = bf16(round(low*low4_scale + high) * scale)   (RNE == jnp.round)
template <bool FULL>
static __device__ __forceinline__ v8bf dq8(v8i8 q, v8i8 h, float ls, float s, float c) {
    v8bf r;
#pragma unroll
    for (int i = 0; i < 8; ++i) {
        float f;
        if (FULL) f = __builtin_rintf((float)q[i] * ls + (float)h[i]) * s;
        else      f = (float)q[i] * c;
        r[i] = (__bf16)f;
    }
    return r;
}

template <bool FULL>
static __device__ __forceinline__ void gemm_body(
    const __bf16* __restrict__ x, const signed char* __restrict__ low,
    const signed char* __restrict__ high, float s, float ls,
    const __bf16* __restrict__ bias, __bf16* __restrict__ out)
{
    const int lane = threadIdx.x & 31;
    const int wave = threadIdx.x >> 5;
    const int lo16 = lane & 15;        // N col (B,C,D) / M row (A)
    const int hi8  = (lane >> 4) * 8;  // K subgroup select / M offset in C/D
    const int nt   = wave >> 1;        // n-tile owned by this wave pair (0..3)
    const int kh   = wave & 1;         // K half (split-K factor 2)
    const int n0   = blockIdx.x * 64 + nt * 16;
    const int kbeg = kh * KHALF;
    const float c  = ls * s;

    // A base: row = lo16 (+ mt*16 per fragment), K packing per ISA 16-bit A layout
    const __bf16*      ap = x    + (size_t)lo16 * IN_F + hi8 + kbeg;
    const signed char* qp = low  + (size_t)(n0 + lo16) * IN_F + hi8 + kbeg;
    const signed char* hp = high + (size_t)(n0 + lo16) * IN_F + hi8 + kbeg;

    v8f acc[4] = {v8f{}, v8f{}, v8f{}, v8f{}};

#pragma unroll 2
    for (int k = 0; k < KHALF; k += 32) {
        // B fragment: 32x16 int8 tile, dequantized exactly ONCE, reused by 4 WMMAs
        v8i8 qa = ldnt8(qp + k), qb = ldnt8(qp + k + 16);
        v8i8 ha = {}, hb = {};
        if (FULL) { ha = ldnt8(hp + k); hb = ldnt8(hp + k + 16); }
        v16bf B = cat16(dq8<FULL>(qa, ha, ls, s, c), dq8<FULL>(qb, hb, ls, s, c));

        // 4 M-tiles against the same B -> 4 independent acc chains (XDL pipelining)
#pragma unroll
        for (int mt = 0; mt < 4; ++mt) {
            const __bf16* a = ap + (size_t)mt * 16 * IN_F + k;
            v16bf A = cat16(*(const v8bf*)a, *(const v8bf*)(a + 16));
            acc[mt] = __builtin_amdgcn_wmma_f32_16x16x32_bf16(
                false, A, false, B, (short)0, acc[mt], false, false);
        }
    }

    // ---- split-K reduction through LDS (workgroup split barrier) ----
    __shared__ float red[4][4][32][8];   // [nt][mt][lane][r] = 16 KB
    if (kh == 1) {
#pragma unroll
        for (int mt = 0; mt < 4; ++mt)
#pragma unroll
            for (int r = 0; r < 8; ++r)
                red[nt][mt][lane][r] = acc[mt][r];   // vectorizes to ds_store_b128
    }
    __syncthreads();
    if (kh == 0) {
        const int n  = n0 + lo16;
        const float bz = (float)bias[n];
#pragma unroll
        for (int mt = 0; mt < 4; ++mt) {
#pragma unroll
            for (int r = 0; r < 8; ++r) {
                const int m = mt * 16 + hi8 + r;     // C/D layout: VGPR r -> row
                const float v = acc[mt][r] + red[nt][mt][lane][r] + bz;
                out[(size_t)m * OUT_F + n] = (__bf16)v;
            }
        }
    }
}

__global__ __launch_bounds__(256)
void GateInformedLinear_24275155157629_kernel(
    const __bf16* __restrict__ x, const signed char* __restrict__ low,
    const signed char* __restrict__ high, const float* __restrict__ scale_p,
    const float* __restrict__ ls_p, const __bf16* __restrict__ bias,
    const int* __restrict__ gate_p, __bf16* __restrict__ out)
{
    const float s  = scale_p[0];
    const float ls = ls_p[0];
    if ((float)gate_p[0] < 0.5f)     // static-in-practice, uniform branch
        gemm_body<false>(x, low, high, s, ls, bias, out);
    else
        gemm_body<true>(x, low, high, s, ls, bias, out);
}

extern "C" void kernel_launch(void* const* d_in, const int* in_sizes, int n_in,
                              void* d_out, int out_size, void* d_ws, size_t ws_size,
                              hipStream_t stream) {
    const __bf16*      x     = (const __bf16*)d_in[0];
    const signed char* low   = (const signed char*)d_in[1];
    const signed char* high  = (const signed char*)d_in[2];
    const float*       scale = (const float*)d_in[3];
    const float*       ls    = (const float*)d_in[4];
    const __bf16*      bias  = (const __bf16*)d_in[5];
    const int*         gate  = (const int*)d_in[6];
    __bf16*            out   = (__bf16*)d_out;

    dim3 grid(OUT_F / 64);   // 172 blocks: 64 N-cols each, full M=64, split-K x2
    dim3 block(256);         // 8 wave32 waves: 4 n-tiles x 2 K-halves
    GateInformedLinear_24275155157629_kernel<<<grid, block, 0, stream>>>(
        x, low, high, scale, ls, bias, gate, out);
}
